// FullNGCFLayer_30940944401034
// MI455X (gfx1250) — matrix operand order repaired and checked by reading, not compiled, not run on hardware
//
#include <hip/hip_runtime.h>

#define N_NODES   100000
#define EMBED_DIM 128
#define LAYER_NUM 3
#define LEAKY     0.2f

typedef __attribute__((ext_vector_type(16))) __bf16 v16bf;
typedef __attribute__((ext_vector_type(8)))  __bf16 v8bf;
typedef __attribute__((ext_vector_type(4)))  __bf16 v4bf;
typedef __attribute__((ext_vector_type(8)))  float  v8f;

union Frag { v16bf v; v8bf h[2]; };

// ---------------------------------------------------------------------------
// Fused dual SpMM: y_LI += li_v * x[col], y_L += l_v * x[col], scatter to row.
// One wave per edge; 32 lanes x float4 = 128 dims. Single gather feeds both
// accumulators (halves gather traffic). Hardware f32 atomics.
// ---------------------------------------------------------------------------
__global__ __launch_bounds__(256) void spmm_dual_kernel(
    const float* __restrict__ li_vals, const float* __restrict__ l_vals,
    const int* __restrict__ rows, const int* __restrict__ cols,
    const float* __restrict__ x,
    float* __restrict__ yLI, float* __restrict__ yL, int nnz)
{
  long long gid = (long long)blockIdx.x * blockDim.x + threadIdx.x;
  int e    = (int)(gid >> 5);
  int lane = (int)(gid & 31);
  if (e >= nnz) return;
  int   r  = rows[e];
  int   c  = cols[e];
  float vi = li_vals[e];
  float vl = l_vals[e];
  const float4 xv = *(const float4*)(x + (size_t)c * EMBED_DIM + lane * 4);
  float* pI = yLI + (size_t)r * EMBED_DIM + lane * 4;
  float* pL = yL  + (size_t)r * EMBED_DIM + lane * 4;
  unsafeAtomicAdd(pI + 0, vi * xv.x);
  unsafeAtomicAdd(pI + 1, vi * xv.y);
  unsafeAtomicAdd(pI + 2, vi * xv.z);
  unsafeAtomicAdd(pI + 3, vi * xv.w);
  unsafeAtomicAdd(pL + 0, vl * xv.x);
  unsafeAtomicAdd(pL + 1, vl * xv.y);
  unsafeAtomicAdd(pL + 2, vl * xv.z);
  unsafeAtomicAdd(pL + 3, vl * xv.w);
}

// ---------------------------------------------------------------------------
// Split W (row-major f32 [128][128]) into bf16 hi/lo, stored TRANSPOSED
// ([n][k]) so B fragments are 16 contiguous bf16 per lane.
// ---------------------------------------------------------------------------
__global__ __launch_bounds__(256) void convert_w_kernel(
    const float* __restrict__ Ws, const float* __restrict__ Wd,
    __bf16* __restrict__ WsHiT, __bf16* __restrict__ WsLoT,
    __bf16* __restrict__ WdHiT, __bf16* __restrict__ WdLoT)
{
  int idx = blockIdx.x * blockDim.x + threadIdx.x;   // 0..16383
  int k = idx >> 7, n = idx & 127;
  float ws = Ws[idx];
  float wd = Wd[idx];
  __bf16 h = (__bf16)ws;
  __bf16 l = (__bf16)(ws - (float)h);
  WsHiT[n * 128 + k] = h;
  WsLoT[n * 128 + k] = l;
  h = (__bf16)wd;
  l = (__bf16)(wd - (float)h);
  WdHiT[n * 128 + k] = h;
  WdLoT[n * 128 + k] = l;
}

// ---------------------------------------------------------------------------
// A1 = LI_side, A2 = L_side * old_ebs; split both into bf16 hi/lo (row-major).
// One thread per float4 (4 dims).
// ---------------------------------------------------------------------------
__global__ __launch_bounds__(256) void convert_a_kernel(
    const float* __restrict__ LI, const float* __restrict__ Lb,
    const float* __restrict__ old_ebs,
    __bf16* __restrict__ A1hi, __bf16* __restrict__ A1lo,
    __bf16* __restrict__ A2hi, __bf16* __restrict__ A2lo)
{
  long long gid = (long long)blockIdx.x * blockDim.x + threadIdx.x;
  if (gid >= (long long)N_NODES * 32) return;
  size_t base = (size_t)gid * 4;
  float4 a = *(const float4*)(LI + base);
  float4 b = *(const float4*)(Lb + base);
  float4 o = *(const float4*)(old_ebs + base);
  float e0 = b.x * o.x, e1 = b.y * o.y, e2 = b.z * o.z, e3 = b.w * o.w;

  __bf16 hx = (__bf16)a.x, hy = (__bf16)a.y, hz = (__bf16)a.z, hw = (__bf16)a.w;
  v4bf h1 = {hx, hy, hz, hw};
  v4bf l1 = {(__bf16)(a.x - (float)hx), (__bf16)(a.y - (float)hy),
             (__bf16)(a.z - (float)hz), (__bf16)(a.w - (float)hw)};
  *(v4bf*)(A1hi + base) = h1;
  *(v4bf*)(A1lo + base) = l1;

  hx = (__bf16)e0; hy = (__bf16)e1; hz = (__bf16)e2; hw = (__bf16)e3;
  v4bf h2 = {hx, hy, hz, hw};
  v4bf l2 = {(__bf16)(e0 - (float)hx), (__bf16)(e1 - (float)hy),
             (__bf16)(e2 - (float)hz), (__bf16)(e3 - (float)hw)};
  *(v4bf*)(A2hi + base) = h2;
  *(v4bf*)(A2lo + base) = l2;
}

// ---------------------------------------------------------------------------
// Fused GEMM: out = leaky_relu(A1 @ Ws + A2 @ Wd), split-bf16 (3 WMMAs per
// split product) with f32 accumulation. One wave per 16-row node tile.
// ---------------------------------------------------------------------------
__device__ __forceinline__ v8f wmma_bf16(v16bf a, v16bf b, v8f c) {
  return __builtin_amdgcn_wmma_f32_16x16x32_bf16(
      /*neg_a=*/false, a, /*neg_b=*/false, b,
      /*c_mod=*/(short)0, c, /*reuse_a=*/false, /*reuse_b=*/false);
}

__global__ __launch_bounds__(256) void gemm_fused_kernel(
    const __bf16* __restrict__ A1hi, const __bf16* __restrict__ A1lo,
    const __bf16* __restrict__ A2hi, const __bf16* __restrict__ A2lo,
    const __bf16* __restrict__ WsHiT, const __bf16* __restrict__ WsLoT,
    const __bf16* __restrict__ WdHiT, const __bf16* __restrict__ WdLoT,
    float* __restrict__ out)
{
  int wave = (int)(((long long)blockIdx.x * blockDim.x + threadIdx.x) >> 5);
  int lane = threadIdx.x & 31;
  if (wave >= N_NODES / 16) return;          // wave-uniform: EXEC all-1s below
  int m  = lane & 15;
  int kh = lane >> 4;

  // A fragments: contiguous K order (same as B) -> lane holds row m, two contiguous
  // 8-element K-runs at Kbase + 16*kh and Kbase + 16*kh + 8.
  size_t arow = (size_t)(wave * 16 + m) * EMBED_DIM;
  Frag a1h[4], a1l[4], a2h[4], a2l[4];
#pragma unroll
  for (int kb = 0; kb < 4; ++kb) {
    size_t b0 = arow + kb * 32 + kh * 16;
    size_t b1 = b0 + 8;
    a1h[kb].h[0] = *(const v8bf*)(A1hi + b0);
    a1h[kb].h[1] = *(const v8bf*)(A1hi + b1);
    a1l[kb].h[0] = *(const v8bf*)(A1lo + b0);
    a1l[kb].h[1] = *(const v8bf*)(A1lo + b1);
    a2h[kb].h[0] = *(const v8bf*)(A2hi + b0);
    a2h[kb].h[1] = *(const v8bf*)(A2hi + b1);
    a2l[kb].h[0] = *(const v8bf*)(A2lo + b0);
    a2l[kb].h[1] = *(const v8bf*)(A2lo + b1);
  }

  for (int nt = 0; nt < 8; ++nt) {
    v8f acc = {};
    int n = nt * 16 + m;                     // B fragment column
#pragma unroll
    for (int kb = 0; kb < 4; ++kb) {
      // B fragment: W^T row n, 16 contiguous K at Kbase + 16*kh.
      size_t wb = (size_t)n * EMBED_DIM + kb * 32 + kh * 16;
      v16bf bsh = *(const v16bf*)(WsHiT + wb);
      v16bf bsl = *(const v16bf*)(WsLoT + wb);
      v16bf bdh = *(const v16bf*)(WdHiT + wb);
      v16bf bdl = *(const v16bf*)(WdLoT + wb);
      // (ah+al)(bh+bl) ~= ah*bh + al*bh + ah*bl  (drop al*bl)
      acc = wmma_bf16(a1h[kb].v, bsh, acc);
      acc = wmma_bf16(a1l[kb].v, bsh, acc);
      acc = wmma_bf16(a1h[kb].v, bsl, acc);
      acc = wmma_bf16(a2h[kb].v, bdh, acc);
      acc = wmma_bf16(a2l[kb].v, bdh, acc);
      acc = wmma_bf16(a2h[kb].v, bdl, acc);
    }
    // C/D layout: VGPR r -> row r (lanes 0-15), row r+8 (lanes 16-31); col = m.
#pragma unroll
    for (int r = 0; r < 8; ++r) {
      float v = acc[r];
      v = (v > 0.0f) ? v : LEAKY * v;
      int orow = wave * 16 + r + kh * 8;
      out[(size_t)orow * EMBED_DIM + nt * 16 + m] = v;
    }
  }
}

// ---------------------------------------------------------------------------
extern "C" void kernel_launch(void* const* d_in, const int* in_sizes, int n_in,
                              void* d_out, int out_size, void* d_ws, size_t ws_size,
                              hipStream_t stream)
{
  const float* initial_ebs = (const float*)d_in[0];
  const float* W_sides     = (const float*)d_in[1];
  const float* W_dots      = (const float*)d_in[2];
  const float* li_vals     = (const float*)d_in[3];
  const float* l_vals      = (const float*)d_in[4];
  const int*   rows        = (const int*)d_in[5];
  const int*   cols        = (const int*)d_in[6];
  float*       out         = (float*)d_out;
  const int    nnz         = in_sizes[3];

  const size_t NE = (size_t)N_NODES * EMBED_DIM;
  char* ws = (char*)d_ws;
  float*  LI    = (float*)ws;   ws += NE * sizeof(float);
  float*  Lb    = (float*)ws;   ws += NE * sizeof(float);
  __bf16* A1hi  = (__bf16*)ws;  ws += NE * sizeof(__bf16);
  __bf16* A1lo  = (__bf16*)ws;  ws += NE * sizeof(__bf16);
  __bf16* A2hi  = (__bf16*)ws;  ws += NE * sizeof(__bf16);
  __bf16* A2lo  = (__bf16*)ws;  ws += NE * sizeof(__bf16);
  __bf16* WsHiT = (__bf16*)ws;  ws += 128 * 128 * sizeof(__bf16);
  __bf16* WsLoT = (__bf16*)ws;  ws += 128 * 128 * sizeof(__bf16);
  __bf16* WdHiT = (__bf16*)ws;  ws += 128 * 128 * sizeof(__bf16);
  __bf16* WdLoT = (__bf16*)ws;  ws += 128 * 128 * sizeof(__bf16);

  const long long spmm_threads = (long long)nnz * 32;
  const int spmm_blocks = (int)((spmm_threads + 255) / 256);
  const int ca_blocks   = (int)(((long long)N_NODES * 32 + 255) / 256);
  const int gemm_blocks = (N_NODES / 16 + 7) / 8;   // 8 waves / block

  const float* old_ebs = initial_ebs;
  for (int layer = 0; layer < LAYER_NUM; ++layer) {
    hipMemsetAsync(LI, 0, NE * sizeof(float), stream);
    hipMemsetAsync(Lb, 0, NE * sizeof(float), stream);

    convert_w_kernel<<<(128 * 128) / 256, 256, 0, stream>>>(
        W_sides + (size_t)layer * 128 * 128,
        W_dots  + (size_t)layer * 128 * 128,
        WsHiT, WsLoT, WdHiT, WdLoT);

    spmm_dual_kernel<<<spmm_blocks, 256, 0, stream>>>(
        li_vals, l_vals, rows, cols, old_ebs, LI, Lb, nnz);

    convert_a_kernel<<<ca_blocks, 256, 0, stream>>>(
        LI, Lb, old_ebs, A1hi, A1lo, A2hi, A2lo);

    float* layer_out = out + (size_t)layer * NE;
    gemm_fused_kernel<<<gemm_blocks, 256, 0, stream>>>(
        A1hi, A1lo, A2hi, A2lo, WsHiT, WsLoT, WdHiT, WdLoT, layer_out);

    old_ebs = layer_out;
  }
}
